// RandLANet_11003706212678
// MI455X (gfx1250) — compile-verified, matrix operand order checked by reading, not run on hardware
//
#include <hip/hip_runtime.h>
#include <cstdint>

// ---------------------------------------------------------------------------
// CDNA5 (gfx1250) RandLA-Net forward: WMMA f16 GEMMs for all 1x1 convs,
// WMMA f32 16x16x4 for xyz Gram matrices, LDS-staged KNN.
// ---------------------------------------------------------------------------

typedef __attribute__((ext_vector_type(16))) _Float16 v16h;
typedef __attribute__((ext_vector_type(8)))  _Float16 v8h;
typedef __attribute__((ext_vector_type(8)))  float    v8f;
typedef __attribute__((ext_vector_type(2)))  float    v2f;

#define EPSI 0.9999950000374997f   // 1/sqrt(1+1e-5)

#if defined(__has_builtin)
#if __has_builtin(__builtin_amdgcn_wmma_f32_16x16x4_f32)
#define HAVE_WMMA_F32X4 1
#endif
#if __has_builtin(__builtin_amdgcn_sqrtf)
#define FAST_SQRT(x) __builtin_amdgcn_sqrtf(x)
#endif
#endif
#ifndef FAST_SQRT
#define FAST_SQRT(x) sqrtf(x)
#endif

// ---------------- xyz pad + norms (level 0) ----------------
__global__ void k_pad_xyz(const float* __restrict__ xyz, float* __restrict__ xyz4,
                          float* __restrict__ nrm, int total) {
  int i = blockIdx.x * blockDim.x + threadIdx.x;
  if (i >= total) return;
  float x = xyz[i * 3 + 0], y = xyz[i * 3 + 1], z = xyz[i * 3 + 2];
  xyz4[i * 4 + 0] = x; xyz4[i * 4 + 1] = y; xyz4[i * 4 + 2] = z; xyz4[i * 4 + 3] = 0.f;
  nrm[i] = x * x + y * y + z * z;
}

// ---------------- fc + bn0 + relu ----------------
__global__ void k_fc(const float* __restrict__ xyz, const float* __restrict__ w,
                     const float* __restrict__ bb, const float* __restrict__ g,
                     const float* __restrict__ bnb, float* __restrict__ out, int total) {
  int i = blockIdx.x * blockDim.x + threadIdx.x;
  if (i >= total) return;
  float x = xyz[i * 3 + 0], y = xyz[i * 3 + 1], z = xyz[i * 3 + 2];
#pragma unroll
  for (int c = 0; c < 8; ++c) {
    float v = x * w[c * 3 + 0] + y * w[c * 3 + 1] + z * w[c * 3 + 2] + bb[c];
    v = v * (g[c] * EPSI) + bnb[c];
    out[i * 8 + c] = fmaxf(v, 0.f);
  }
}

// ---------------- density: sum of pairwise distances per row ----------------
// one wave per 16-row tile; Gram via v_wmma_f32_16x16x4_f32
__global__ void k_density(const float* __restrict__ xyz4, const float* __restrict__ nrm,
                          float* __restrict__ sumd, int M) {
  int wave = (blockIdx.x * blockDim.x + threadIdx.x) >> 5;
  int lane = threadIdx.x & 31;
  int tilesR = M >> 4;
  int b = wave / tilesR;
  int tr = wave % tilesR;
  if (b >= 2) return;
  const float* X = xyz4 + (size_t)b * M * 4;
  const float* Nn = nrm + (size_t)b * M;
  int half = lane >> 4;
  int col = lane & 15;
  int r0 = tr * 16;
  float rn[8];
#pragma unroll
  for (int j = 0; j < 8; ++j) rn[j] = Nn[r0 + half * 8 + j];
  float partial[8];
#pragma unroll
  for (int j = 0; j < 8; ++j) partial[j] = 0.f;

#ifdef HAVE_WMMA_F32X4
  v2f afr;
  { int kb = half * 2;
    afr[0] = X[(r0 + col) * 4 + kb + 0];
    afr[1] = X[(r0 + col) * 4 + kb + 1]; }
  for (int c0 = 0; c0 < M; c0 += 16) {
    v2f bfr;
    { int kb = half * 2;
      bfr[0] = X[(c0 + col) * 4 + kb + 0];
      bfr[1] = X[(c0 + col) * 4 + kb + 1]; }
    v8f gacc = {};
    gacc = __builtin_amdgcn_wmma_f32_16x16x4_f32(false, afr, false, bfr,
                                                 (short)0, gacc, false, false);
    float cn = Nn[c0 + col];
#pragma unroll
    for (int j = 0; j < 8; ++j) {
      float d2 = rn[j] + cn - 2.f * gacc[j];
      partial[j] += FAST_SQRT(fmaxf(d2, 0.f) + 1e-12f);
    }
  }
#else
  // scalar fallback: lane handles one (row-half, col) strip
  for (int c0 = 0; c0 < M; c0 += 16) {
    int cc = c0 + col;
    float cx = X[cc * 4 + 0], cy = X[cc * 4 + 1], cz = X[cc * 4 + 2];
#pragma unroll
    for (int j = 0; j < 8; ++j) {
      int rr = r0 + half * 8 + j;
      float dx = X[rr * 4 + 0] - cx, dy = X[rr * 4 + 1] - cy, dz = X[rr * 4 + 2] - cz;
      partial[j] += FAST_SQRT(dx * dx + dy * dy + dz * dz + 1e-12f);
    }
  }
#endif
  // reduce over the 16 lanes of each half (columns), then lane0/lane16 write
#pragma unroll
  for (int j = 0; j < 8; ++j) {
#pragma unroll
    for (int mask = 1; mask < 16; mask <<= 1)
      partial[j] += __shfl_xor(partial[j], mask, 32);
  }
  if (col == 0) {
#pragma unroll
    for (int j = 0; j < 8; ++j)
      sumd[(size_t)b * M + r0 + half * 8 + j] = partial[j];
  }
}

// ---------------- sampling: keep m smallest sumd (== largest density) -------
__global__ void k_mark(const float* __restrict__ sumd, int M, int m, int* __restrict__ kept) {
  int idx = blockIdx.x * blockDim.x + threadIdx.x;
  if (idx >= 2 * M) return;
  int i = idx % M, b = idx / M;
  const float* S = sumd + (size_t)b * M;
  float v = S[i];
  int rank = 0;
  for (int j = 0; j < M; ++j) {
    float u = S[j];
    rank += (u < v) || (u == v && j < i);
  }
  kept[idx] = (rank < m) ? 1 : 0;
}

__global__ void k_compact(const int* __restrict__ kept, int M, int m, int* __restrict__ sidx) {
  int idx = blockIdx.x * blockDim.x + threadIdx.x;
  if (idx >= 2 * M) return;
  if (!kept[idx]) return;
  int i = idx % M, b = idx / M;
  const int* K = kept + (size_t)b * M;
  int pos = 0;
  for (int j = 0; j < i; ++j) pos += K[j];
  if (pos < m) sidx[(size_t)b * m + pos] = i;
}

__global__ void k_gather_sampled(const float* __restrict__ xyz4in, const float* __restrict__ featin,
                                 const int* __restrict__ sidx,
                                 float* __restrict__ xyz4out, float* __restrict__ nrmout,
                                 float* __restrict__ featout, int Min, int m, int C) {
  int idx = blockIdx.x * blockDim.x + threadIdx.x;
  if (idx >= 2 * m) return;
  int b = idx / m;
  int s = sidx[idx];
  const float* p = xyz4in + ((size_t)b * Min + s) * 4;
  float x = p[0], y = p[1], z = p[2];
  xyz4out[idx * 4 + 0] = x; xyz4out[idx * 4 + 1] = y;
  xyz4out[idx * 4 + 2] = z; xyz4out[idx * 4 + 3] = 0.f;
  nrmout[idx] = x * x + y * y + z * z;
  const float* fi = featin + ((size_t)b * Min + s) * C;
  float* fo = featout + (size_t)idx * C;
  for (int c = 0; c < C; ++c) fo[c] = fi[c];
}

// ---------------- density-weighted KNN (LDS-staged brute force) -------------
template <int K2, int KOUT>
__global__ void k_knn(const float* __restrict__ xyz4, int* __restrict__ out, int m) {
  __shared__ float tx[64], ty[64], tz[64];
  int b = blockIdx.y;
  int q = blockIdx.x * 64 + (int)threadIdx.x;
  bool valid = q < m;
  const float* X = xyz4 + (size_t)b * m * 4;
  float qx = 0.f, qy = 0.f, qz = 0.f;
  if (valid) { qx = X[q * 4 + 0]; qy = X[q * 4 + 1]; qz = X[q * 4 + 2]; }
  float bd[K2]; int bi[K2];
#pragma unroll
  for (int t = 0; t < K2; ++t) { bd[t] = 3.4e38f; bi[t] = 0; }
  for (int c0 = 0; c0 < m; c0 += 64) {
    int c = c0 + (int)threadIdx.x;
    if (c < m) { tx[threadIdx.x] = X[c * 4 + 0]; ty[threadIdx.x] = X[c * 4 + 1]; tz[threadIdx.x] = X[c * 4 + 2]; }
    __syncthreads();
    int lim = min(64, m - c0);
    for (int j = 0; j < lim; ++j) {
      float dx = tx[j] - qx, dy = ty[j] - qy, dz = tz[j] - qz;
      float d2 = dx * dx + dy * dy + dz * dz;
      if (valid && d2 < bd[K2 - 1]) {
        float cd = d2; int ci = c0 + j;
#pragma unroll
        for (int t = 0; t < K2; ++t) {
          if (cd < bd[t]) {
            float td = bd[t]; int ti = bi[t];
            bd[t] = cd; bi[t] = ci; cd = td; ci = ti;
          }
        }
      }
    }
    __syncthreads();
  }
  if (!valid) return;
  // local stats (two-pass gather to limit registers)
  float mx = 0.f, my = 0.f, mz = 0.f;
#pragma unroll
  for (int t = 0; t < K2; ++t) {
    const float* p = X + (size_t)bi[t] * 4;
    mx += p[0]; my += p[1]; mz += p[2];
  }
  mx *= (1.f / K2); my *= (1.f / K2); mz *= (1.f / K2);
  float vx = 0.f, vy = 0.f, vz = 0.f;
#pragma unroll
  for (int t = 0; t < K2; ++t) {
    const float* p = X + (size_t)bi[t] * 4;
    float dx = p[0] - mx, dy = p[1] - my, dz = p[2] - mz;
    vx += dx * dx; vy += dy * dy; vz += dz * dz;
  }
  float inv = 1.f / (float)(K2 - 1);
  float denom = ((FAST_SQRT(vx * inv) + 1e-6f) + (FAST_SQRT(vy * inv) + 1e-6f) +
                 (FAST_SQRT(vz * inv) + 1e-6f)) * (1.f / 3.f);
  float key[K2];
#pragma unroll
  for (int t = 0; t < K2; ++t) {
    const float* p = X + (size_t)bi[t] * 4;
    float dx = p[0] - mx, dy = p[1] - my, dz = p[2] - mz;
    float ss = dx * dx + dy * dy + dz * dz;
    key[t] = bd[t] * __expf(-ss / denom);
  }
  // stable selection of KOUT smallest keys
#pragma unroll
  for (int o = 0; o < KOUT; ++o) {
    float bk = 3.5e38f; int bidx = 0; int bt = -1;
#pragma unroll
    for (int t = 0; t < K2; ++t) {
      if (key[t] < bk) { bk = key[t]; bidx = bi[t]; bt = t; }
    }
    out[((size_t)b * m + q) * KOUT + o] = bidx;
#pragma unroll
    for (int t = 0; t < K2; ++t) if (t == bt) key[t] = 3.5e38f;
  }
}

// ---------------- gather + concat [base, fc, xc] into f16 rows --------------
__global__ void k_gather_concat(const float* __restrict__ xyz4, const float* __restrict__ feat,
                                const int* __restrict__ knn, _Float16* __restrict__ H,
                                int m, int K, int C, int Kpad) {
  int idx = blockIdx.x * blockDim.x + threadIdx.x;
  int total = 2 * m * K;
  if (idx >= total) return;
  int i = (idx / K) % m;
  int b = idx / (K * m);
  const float* Xb = xyz4 + (size_t)b * m * 4;
  const float* Fb = feat + (size_t)b * m * C;
  int j = knn[idx];
  _Float16* row = H + (size_t)idx * Kpad;
  for (int c = 0; c < C; ++c) {
    float fi = Fb[(size_t)i * C + c], fj = Fb[(size_t)j * C + c];
    row[c] = (_Float16)fi;
    row[C + c] = (_Float16)(fj - fi);
  }
  row[2 * C + 0] = (_Float16)(Xb[j * 4 + 0] - Xb[i * 4 + 0]);
  row[2 * C + 1] = (_Float16)(Xb[j * 4 + 1] - Xb[i * 4 + 1]);
  row[2 * C + 2] = (_Float16)(Xb[j * 4 + 2] - Xb[i * 4 + 2]);
  for (int c = 2 * C + 3; c < Kpad; ++c) row[c] = (_Float16)0.f;
}

// ---------------- generic WMMA f16 GEMM: O = act((A @ Wt) * s + t) ------------
// Each wave owns a 16x32 output tile (two N-tiles share one A fragment),
// falling back to 16x16 when Np == 16.
__device__ __forceinline__ v16h load_frag(const _Float16* base, int ld, int lane) {
  int r = lane & 15;
  int hk = (lane >> 4) * 8;
  const _Float16* p = base + (size_t)r * ld + hk;
  v8h lo = *(const v8h*)(p);
  v8h hi = *(const v8h*)(p + 16);
  v16h a;
#pragma unroll
  for (int t = 0; t < 8; ++t) { a[t] = lo[t]; a[t + 8] = hi[t]; }
  return a;
}

__global__ __launch_bounds__(256)
void k_gemm_wmma(const _Float16* __restrict__ A, int lda,
                 const _Float16* __restrict__ W, int ldw,
                 _Float16* __restrict__ O, int ldo,
                 const float* __restrict__ sc, const float* __restrict__ tc,
                 int Mr, int Kp, int Np, int relu) {
  int wave = (blockIdx.x * blockDim.x + threadIdx.x) >> 5;
  int lane = threadIdx.x & 31;
  int tilesN = Np >> 4;
  int pairsN = (tilesN + 1) >> 1;
  int tm = wave / pairsN;
  int tn = (wave % pairsN) * 2;
  if (tm * 16 >= Mr) return;
  bool two = (tn + 1) < tilesN;           // wave-uniform
  const _Float16* Ab  = A + (size_t)(tm * 16) * lda;
  const _Float16* Wb0 = W + (size_t)(tn * 16) * ldw;
  const _Float16* Wb1 = Wb0 + (size_t)16 * ldw;
  v8f acc0 = {}, acc1 = {};
  for (int kk = 0; kk < Kp; kk += 32) {
    __builtin_prefetch(Ab + kk + 64, 0, 1);
    v16h a  = load_frag(Ab + kk, lda, lane);
    v16h b0 = load_frag(Wb0 + kk, ldw, lane);
    acc0 = __builtin_amdgcn_wmma_f32_16x16x32_f16(false, a, false, b0,
                                                  (short)0, acc0, false, false);
    if (two) {
      v16h b1 = load_frag(Wb1 + kk, ldw, lane);
      acc1 = __builtin_amdgcn_wmma_f32_16x16x32_f16(false, a, false, b1,
                                                    (short)0, acc1, false, false);
    }
  }
  int rbase = tm * 16 + ((lane >> 4) << 3);
  {
    int n = tn * 16 + (lane & 15);
    float s = sc[n], t = tc[n];
#pragma unroll
    for (int j = 0; j < 8; ++j) {
      float y = acc0[j] * s + t;
      if (relu) y = fmaxf(y, 0.f);
      O[(size_t)(rbase + j) * ldo + n] = (_Float16)y;
    }
  }
  if (two) {
    int n = (tn + 1) * 16 + (lane & 15);
    float s = sc[n], t = tc[n];
#pragma unroll
    for (int j = 0; j < 8; ++j) {
      float y = acc1[j] * s + t;
      if (relu) y = fmaxf(y, 0.f);
      O[(size_t)(rbase + j) * ldo + n] = (_Float16)y;
    }
  }
}

// ---------------- weight prep: pad to [Np,Kp] f16 + epilogue s/t ------------
__global__ void k_prep_w(const float* __restrict__ w, const float* __restrict__ g,
                         const float* __restrict__ bb, int Cout, int Cin,
                         _Float16* __restrict__ w16, float* __restrict__ s,
                         float* __restrict__ t, int Np, int Kp, int useBN) {
  int idx = blockIdx.x * blockDim.x + threadIdx.x;
  if (idx >= Np * Kp) return;
  int c = idx % Kp, o = idx / Kp;
  float v = (o < Cout && c < Cin) ? w[(size_t)o * Cin + c] : 0.f;
  w16[idx] = (_Float16)v;
  if (c == 0) {
    s[o] = (o < Cout) ? (useBN ? g[o] * EPSI : 1.f) : 0.f;
    t[o] = (o < Cout) ? bb[o] : 0.f;
  }
}

// ---------------- max-pool over K neighbors, f16 -> f32 ---------------------
__global__ void k_maxpool(const _Float16* __restrict__ Hin, int ldh,
                          float* __restrict__ out, int m, int K, int C) {
  int idx = blockIdx.x * blockDim.x + threadIdx.x;
  int total = 2 * m * C;
  if (idx >= total) return;
  int c = idx % C;
  int i = (idx / C) % m;
  int b = idx / (C * m);
  const _Float16* base = Hin + ((size_t)(b * m + i) * K) * ldh + c;
  float best = -3.4e38f;
  for (int t = 0; t < K; ++t) best = fmaxf(best, (float)base[(size_t)t * ldh]);
  out[(size_t)(b * m + i) * C + c] = best;
}

// ---------------- f32 rows -> f16 rows (with col offset) --------------------
__global__ void k_f32f16(const float* __restrict__ in, int C, int rows,
                         _Float16* __restrict__ out, int ldo, int colOff) {
  int idx = blockIdx.x * blockDim.x + threadIdx.x;
  if (idx >= rows * C) return;
  int c = idx % C, r = idx / C;
  out[(size_t)r * ldo + colOff + c] = (_Float16)in[(size_t)r * C + c];
}

// ---------------- linear interpolation along points -------------------------
__global__ void k_interp(const _Float16* __restrict__ in, int n_in, int C, int ldi,
                         _Float16* __restrict__ out, int n_out, int ldo, int colOff) {
  int idx = blockIdx.x * blockDim.x + threadIdx.x;
  int total = 2 * n_out * C;
  if (idx >= total) return;
  int c = idx % C;
  int j = (idx / C) % n_out;
  int b = idx / (C * n_out);
  float pos = (j + 0.5f) * ((float)n_in / (float)n_out) - 0.5f;
  pos = fminf(fmaxf(pos, 0.f), (float)(n_in - 1));
  int lo = (int)floorf(pos);
  int hi = min(lo + 1, n_in - 1);
  float w = pos - (float)lo;
  float v = (float)in[((size_t)b * n_in + lo) * ldi + c] * (1.f - w) +
            (float)in[((size_t)b * n_in + hi) * ldi + c] * w;
  out[((size_t)b * n_out + j) * ldo + colOff + c] = (_Float16)v;
}

// ---------------- final [B,N,Cp] f16 -> [B,5,N] f32 -------------------------
__global__ void k_out_tr(const _Float16* __restrict__ tmp, float* __restrict__ out, int N) {
  int idx = blockIdx.x * blockDim.x + threadIdx.x;
  if (idx >= 2 * 5 * N) return;
  int n = idx % N;
  int c = (idx / N) % 5;
  int b = idx / (5 * N);
  out[idx] = (float)tmp[((size_t)b * N + n) * 16 + c];
}

// ===========================================================================
// host orchestration
// ===========================================================================
namespace {
struct Arena {
  char* base; size_t off;
  void* get(size_t bytes) {
    off = (off + 255) & ~(size_t)255;
    void* r = base + off;
    off += bytes;
    return r;
  }
};
struct PW { _Float16* w; float* s; float* t; };
inline int ru(int x, int r) { return (x + r - 1) / r * r; }
inline unsigned nb(long long n, int t) { return (unsigned)((n + t - 1) / t); }
}  // namespace

extern "C" void kernel_launch(void* const* d_in, const int* in_sizes, int n_in,
                              void* d_out, int out_size, void* d_ws, size_t ws_size,
                              hipStream_t stream) {
  (void)in_sizes; (void)n_in; (void)out_size; (void)ws_size;
  const int N = 8192;
  static const int Ms[5] = {8192, 2048, 512, 128, 32};
  struct ECfg { int cin, cout, k, k2, c2, ci; };
  static const ECfg EC[4] = {
      {8, 16, 16, 32, 8, 19}, {16, 64, 8, 16, 32, 35},
      {64, 128, 5, 10, 64, 131}, {128, 256, 4, 8, 128, 259}};
  struct UCfg { int cp, cs, co; };
  static const UCfg UC[4] = {{256, 128, 128}, {128, 64, 64}, {64, 16, 32}, {32, 8, 32}};

  // ---- inputs (setup_inputs dict order, params nested depth-first) ----
  int ip = 0;
  const float* xyz  = (const float*)d_in[ip++];
  const float* fc_w = (const float*)d_in[ip++];
  const float* fc_b = (const float*)d_in[ip++];
  const float* bn0g = (const float*)d_in[ip++];
  const float* bn0b = (const float*)d_in[ip++];
  const float* encw[4][9];
  for (int i = 0; i < 4; ++i)
    for (int j = 0; j < 9; ++j) encw[i][j] = (const float*)d_in[ip++];
  const float* upw[4][6];
  for (int i = 0; i < 4; ++i)
    for (int j = 0; j < 6; ++j) upw[i][j] = (const float*)d_in[ip++];
  const float* seg_w1 = (const float*)d_in[ip++];
  const float* seg_g  = (const float*)d_in[ip++];
  const float* seg_b  = (const float*)d_in[ip++];
  const float* seg_w2 = (const float*)d_in[ip++];
  const float* seg_b2 = (const float*)d_in[ip++];

  // ---- scratch arena ----
  Arena ar{(char*)d_ws, 0};
  float* xyz4[5]; float* nrm[5]; float* feat[5];
  for (int l = 0; l < 5; ++l) {
    xyz4[l] = (float*)ar.get((size_t)2 * Ms[l] * 4 * sizeof(float));
    nrm[l]  = (float*)ar.get((size_t)2 * Ms[l] * sizeof(float));
  }
  feat[0] = (float*)ar.get((size_t)2 * N * 8 * sizeof(float));
  for (int l = 0; l < 4; ++l)
    feat[l + 1] = (float*)ar.get((size_t)2 * Ms[l + 1] * EC[l].cout * sizeof(float));
  float* featS = (float*)ar.get((size_t)2 * 2048 * 64 * sizeof(float));
  float* sumd  = (float*)ar.get((size_t)2 * N * sizeof(float));
  int*   kept  = (int*)ar.get((size_t)2 * N * sizeof(int));
  int*   sidx  = (int*)ar.get((size_t)2 * 2048 * sizeof(int));
  int*   knn   = (int*)ar.get((size_t)2 * 2048 * 16 * sizeof(int));
  _Float16* Hb = (_Float16*)ar.get((size_t)4 * 1024 * 1024 + 65536);
  _Float16* T1 = (_Float16*)ar.get((size_t)4 * 1024 * 1024 + 65536);
  _Float16* T2 = (_Float16*)ar.get((size_t)4 * 1024 * 1024 + 65536);
  _Float16* T3 = (_Float16*)ar.get((size_t)2 * 1024 * 1024 + 65536);

  auto prep = [&](const float* w, const float* g, const float* bb,
                  int Cout, int Cin, int useBN) -> PW {
    int Np = ru(Cout, 16), Kp = ru(Cin, 32);
    PW r;
    r.w = (_Float16*)ar.get((size_t)Np * Kp * sizeof(_Float16));
    r.s = (float*)ar.get((size_t)Np * sizeof(float));
    r.t = (float*)ar.get((size_t)Np * sizeof(float));
    int n = Np * Kp;
    k_prep_w<<<nb(n, 256), 256, 0, stream>>>(w, g, bb, Cout, Cin, r.w, r.s, r.t, Np, Kp, useBN);
    return r;
  };
  auto gemm = [&](const _Float16* A, int lda, const PW& W, int Kp,
                  _Float16* O, int ldo, int Mr, int Np, int relu) {
    int tilesM = Mr / 16;
    int pairsN = ((Np >> 4) + 1) >> 1;
    int waves = tilesM * pairsN;
    int blocks = (waves + 7) / 8;
    k_gemm_wmma<<<blocks, 256, 0, stream>>>(A, lda, W.w, Kp, O, ldo, W.s, W.t, Mr, Kp, Np, relu);
  };

  // ---- prep all weights ----
  PW ew[4][3], uw[4][2], sw1, sw2;
  for (int l = 0; l < 4; ++l) {
    ew[l][0] = prep(encw[l][0], encw[l][1], encw[l][2], EC[l].c2,   EC[l].ci, 1);
    ew[l][1] = prep(encw[l][3], encw[l][4], encw[l][5], EC[l].c2,   EC[l].c2, 1);
    ew[l][2] = prep(encw[l][6], encw[l][7], encw[l][8], EC[l].cout, EC[l].c2, 1);
  }
  for (int i = 0; i < 4; ++i) {
    uw[i][0] = prep(upw[i][0], upw[i][1], upw[i][2], UC[i].co, UC[i].cp + UC[i].cs, 1);
    uw[i][1] = prep(upw[i][3], upw[i][4], upw[i][5], UC[i].co, UC[i].co, 1);
  }
  sw1 = prep(seg_w1, seg_g, seg_b, 64, 32, 1);
  sw2 = prep(seg_w2, nullptr, seg_b2, 5, 64, 0);

  // ---- level 0 features ----
  k_pad_xyz<<<nb(2 * N, 256), 256, 0, stream>>>(xyz, xyz4[0], nrm[0], 2 * N);
  k_fc<<<nb(2 * N, 256), 256, 0, stream>>>(xyz, fc_w, fc_b, bn0g, bn0b, feat[0], 2 * N);

  // ---- encoder ----
  int Cl = 8;
  for (int l = 0; l < 4; ++l) {
    int M = Ms[l], m = Ms[l + 1], K = EC[l].k;
    // density -> sample
    { int waves = 2 * M / 16;
      k_density<<<nb((long long)waves * 32, 128), 128, 0, stream>>>(xyz4[l], nrm[l], sumd, M); }
    k_mark<<<nb(2 * M, 256), 256, 0, stream>>>(sumd, M, m, kept);
    k_compact<<<nb(2 * M, 256), 256, 0, stream>>>(kept, M, m, sidx);
    k_gather_sampled<<<nb(2 * m, 256), 256, 0, stream>>>(
        xyz4[l], feat[l], sidx, xyz4[l + 1], nrm[l + 1], featS, M, m, Cl);
    // knn on sampled set
    dim3 kg(nb(m, 64), 2);
    switch (l) {
      case 0: k_knn<32, 16><<<kg, 64, 0, stream>>>(xyz4[l + 1], knn, m); break;
      case 1: k_knn<16, 8><<<kg, 64, 0, stream>>>(xyz4[l + 1], knn, m); break;
      case 2: k_knn<10, 5><<<kg, 64, 0, stream>>>(xyz4[l + 1], knn, m); break;
      default: k_knn<8, 4><<<kg, 64, 0, stream>>>(xyz4[l + 1], knn, m); break;
    }
    // gather + concat -> H
    int Kp1 = ru(EC[l].ci, 32);
    int rows = 2 * m * K;
    k_gather_concat<<<nb(rows, 256), 256, 0, stream>>>(xyz4[l + 1], featS, knn, Hb, m, K, Cl, Kp1);
    // 3x (1x1 conv + bn + relu) via WMMA
    int Np1 = ru(EC[l].c2, 16), st1 = ru(EC[l].c2, 32);
    int Np3 = ru(EC[l].cout, 16);
    if (st1 > Np1) {
      hipMemsetAsync(T1, 0, (size_t)rows * st1 * sizeof(_Float16), stream);
      hipMemsetAsync(T2, 0, (size_t)rows * st1 * sizeof(_Float16), stream);
    }
    gemm(Hb, Kp1, ew[l][0], Kp1, T1, st1, rows, Np1, 1);
    gemm(T1, st1, ew[l][1], st1, T2, st1, rows, Np1, 1);
    gemm(T2, st1, ew[l][2], st1, T3, Np3, rows, Np3, 1);
    // max-pool over neighbors
    k_maxpool<<<nb((long long)2 * m * EC[l].cout, 256), 256, 0, stream>>>(
        T3, Np3, feat[l + 1], m, K, EC[l].cout);
    Cl = EC[l].cout;
  }

  // ---- decoder ----
  // dec0 = feat4 as f16 [2,32,256] in T2
  k_f32f16<<<nb(2 * 32 * 256, 256), 256, 0, stream>>>(feat[4], 256, 2 * 32, T2, 256, 0);
  int decC = 256, decN = 32;
  for (int i = 0; i < 4; ++i) {
    int cp = UC[i].cp, cs = UC[i].cs, co = UC[i].co;
    int nsk = Ms[3 - i];
    const float* skipF = feat[3 - i];
    int Kc = cp + cs, ld = ru(Kc, 32);
    int rows = 2 * nsk;
    if (ld > Kc) hipMemsetAsync(Hb, 0, (size_t)rows * ld * sizeof(_Float16), stream);
    k_interp<<<nb((long long)rows * cp, 256), 256, 0, stream>>>(T2, decN, cp, decC, Hb, nsk, ld, 0);
    k_f32f16<<<nb((long long)rows * cs, 256), 256, 0, stream>>>(skipF, cs, rows, Hb, ld, cp);
    gemm(Hb, ld, uw[i][0], ld, T1, co, rows, co, 1);
    gemm(T1, co, uw[i][1], co, T2, co, rows, co, 1);
    decC = co; decN = nsk;
  }

  // ---- segmentation head ----
  int rows = 2 * N;
  gemm(T2, 32, sw1, 32, T1, 64, rows, 64, 1);   // seg1 + bn + relu
  gemm(T1, 64, sw2, 64, T3, 16, rows, 16, 0);   // seg2 + bias (no relu)
  k_out_tr<<<nb(2 * 5 * N, 256), 256, 0, stream>>>(T3, (float*)d_out, N);
}
